// MultiHeadedTrilinearAttention_429496729639
// MI455X (gfx1250) — matrix operand order repaired and checked by reading, not compiled, not run on hardware
//
#include <hip/hip_runtime.h>
#include <hip/hip_bf16.h>

#define DIM   768
#define HEADS 12
#define HW    64
#define BB    32
#define NVv   100
#define NQq   32
#define NAa   32

typedef __attribute__((ext_vector_type(16))) __bf16       v16bf;
typedef __attribute__((ext_vector_type(8)))  float        v8f;
typedef __attribute__((ext_vector_type(8)))  unsigned int v8u;
typedef __attribute__((ext_vector_type(4)))  unsigned int v4u;

// ---------- bf16 helpers (bit-level, no __bf16 arithmetic) ----------
__device__ __forceinline__ unsigned short f2bf(float f) {
  unsigned int u = __builtin_bit_cast(unsigned int, f);
  u += 0x7FFFu + ((u >> 16) & 1u);           // round-to-nearest-even
  return (unsigned short)(u >> 16);
}
__device__ __forceinline__ float bf2f(unsigned short h) {
  return __builtin_bit_cast(float, ((unsigned int)h) << 16);
}

// elementwise a*b on bf16 pairs packed in v8u
__device__ __forceinline__ v8u bfmul(v8u a, v8u b) {
  v8u r;
#pragma unroll
  for (int i = 0; i < 8; ++i) {
    unsigned ua = a[i], ub = b[i];
    float lo = bf2f((unsigned short)(ua & 0xFFFFu)) * bf2f((unsigned short)(ub & 0xFFFFu));
    float hi = bf2f((unsigned short)(ua >> 16))     * bf2f((unsigned short)(ub >> 16));
    r[i] = (unsigned)f2bf(lo) | ((unsigned)f2bf(hi) << 16);
  }
  return r;
}
// elementwise a*s (s = power of two -> exact in bf16)
__device__ __forceinline__ v8u bfscale(v8u a, float s) {
  v8u r;
#pragma unroll
  for (int i = 0; i < 8; ++i) {
    unsigned ua = a[i];
    float lo = bf2f((unsigned short)(ua & 0xFFFFu)) * s;
    float hi = bf2f((unsigned short)(ua >> 16))     * s;
    r[i] = (unsigned)f2bf(lo) | ((unsigned)f2bf(hi) << 16);
  }
  return r;
}

// ---------- WMMA wrapper ----------
__device__ __forceinline__ v8f wmma_bf16(v8u a, v8u b, v8f c) {
  return __builtin_amdgcn_wmma_f32_16x16x32_bf16(
      false, __builtin_bit_cast(v16bf, a),
      false, __builtin_bit_cast(v16bf, b),
      (short)0, c, false, false);
}

// ---------- fragment loaders (contiguous k order: lane half h holds k 16h..16h+15) ----------
// A fragment 16x32 bf16 from row-major [rows x ld]; rows r0..r0+15, cols k0..k0+31
__device__ __forceinline__ v8u load_afrag(const unsigned short* p, int ld, int r0, int k0, int lane) {
  int half = lane >> 4;
  const unsigned short* base = p + (size_t)(r0 + (lane & 15)) * ld + k0 + half * 16;
  v4u x = *(const v4u*)base;          // K offsets +0..7 (lo half) / +16..23 (hi half)
  v4u y = *(const v4u*)(base + 8);    // K offsets +8..15 / +24..31
  v8u r;
  r[0]=x[0]; r[1]=x[1]; r[2]=x[2]; r[3]=x[3];
  r[4]=y[0]; r[5]=y[1]; r[6]=y[2]; r[7]=y[3];
  return r;
}
// A fragment where every M-row is the same single row (broadcast of vh)
__device__ __forceinline__ v8u load_afrag_bcast(const unsigned short* rowp, int k0, int lane) {
  int half = lane >> 4;
  const unsigned short* base = rowp + k0 + half * 16;
  v4u x = *(const v4u*)base;
  v4u y = *(const v4u*)(base + 8);
  v8u r;
  r[0]=x[0]; r[1]=x[1]; r[2]=x[2]; r[3]=x[3];
  r[4]=y[0]; r[5]=y[1]; r[6]=y[2]; r[7]=y[3];
  return r;
}
// B fragment 32x16 with B[k][n] = rows[n][k]; rows is N x K row-major slice (ah^T)
__device__ __forceinline__ v8u load_bfrag_rows(const unsigned short* p, int ld, int n0, int k0, int lane) {
  int half = lane >> 4;
  const unsigned short* base = p + (size_t)(n0 + (lane & 15)) * ld + k0 + half * 16;
  v4u x = *(const v4u*)base;
  v4u y = *(const v4u*)(base + 8);
  v8u r;
  r[0]=x[0]; r[1]=x[1]; r[2]=x[2]; r[3]=x[3];
  r[4]=y[0]; r[5]=y[1]; r[6]=y[2]; r[7]=y[3];
  return r;
}
// B fragment from pre-packed weight tile (512 bf16, lane-major: elem = tile[lane*16+e])
__device__ __forceinline__ v8u load_bfrag_packed(const unsigned short* tile, int lane) {
  const unsigned short* base = tile + lane * 16;
  v4u x = *(const v4u*)base;
  v4u y = *(const v4u*)(base + 8);
  v8u r;
  r[0]=x[0]; r[1]=x[1]; r[2]=x[2]; r[3]=x[3];
  r[4]=y[0]; r[5]=y[1]; r[6]=y[2]; r[7]=y[3];
  return r;
}

// ---------- ordered-float <-> uint key (for atomic max over floats) ----------
__device__ __forceinline__ unsigned int fkey(float f) {
  unsigned int u = __builtin_bit_cast(unsigned int, f);
  return (u & 0x80000000u) ? ~u : (u | 0x80000000u);
}
__device__ __forceinline__ float unfkey(unsigned int k) {
  unsigned int u = (k & 0x80000000u) ? (k ^ 0x80000000u) : ~k;
  return __builtin_bit_cast(float, u);
}

// ================= kernels =================

struct PackArgs { const float* W[6]; unsigned short* dst; };

// Pack W (768x768 f32 row-major) into B-fragment lane layout:
// packed[mat][kt*48+nt][lane*16+e] = bf16( W[kt*32 + (lane>=16)*16 + e][nt*16 + lane%16] )
__global__ void pack_weights_kernel(PackArgs args) {
  int gw   = (blockIdx.x * blockDim.x + threadIdx.x) >> 5;
  int lane = threadIdx.x & 31;
  if (gw >= 6 * 24 * 48) return;
  int mat = gw / (24 * 48);
  int t   = gw % (24 * 48);
  int kt  = t / 48, nt = t % 48;
  const float* W = args.W[mat];
  unsigned short* dst = args.dst + (size_t)mat * DIM * DIM + (size_t)t * 512 + lane * 16;
  int kbase = kt * 32 + ((lane >> 4) << 4);
  int n     = nt * 16 + (lane & 15);
  unsigned int out[8];
#pragma unroll
  for (int i = 0; i < 8; ++i) {
    unsigned short lo = f2bf(W[(size_t)(kbase + 2*i    ) * DIM + n]);
    unsigned short hi = f2bf(W[(size_t)(kbase + 2*i + 1) * DIM + n]);
    out[i] = (unsigned)lo | ((unsigned)hi << 16);
  }
  v4u* d = (v4u*)dst;
  v4u t0; t0[0]=out[0]; t0[1]=out[1]; t0[2]=out[2]; t0[3]=out[3];
  v4u t1; t1[0]=out[4]; t1[1]=out[5]; t1[2]=out[6]; t1[3]=out[7];
  d[0] = t0; d[1] = t1;
}

__global__ void cvt_f32_bf16_kernel(const float* __restrict__ src, unsigned short* __restrict__ dst, int n) {
  int i = blockIdx.x * blockDim.x + threadIdx.x;
  if (i < n) dst[i] = f2bf(src[i]);
}

// C(bf16, MxN) = A(bf16, MxK row-major) @ Wpacked + bias ; N=K=768, 32x32 tile per wave
__global__ void gemm_bf16_kernel(const unsigned short* __restrict__ A,
                                 const unsigned short* __restrict__ Wp,
                                 const float* __restrict__ bias,
                                 unsigned short* __restrict__ C, int M) {
  int gw   = (blockIdx.x * blockDim.x + threadIdx.x) >> 5;
  int lane = threadIdx.x & 31;
  int mtiles = M >> 5;
  if (gw >= mtiles * 24) return;
  int mt = gw / 24, nt = gw % 24;
  int m0 = mt * 32, n0 = nt * 32;
  v8f z = {};
  v8f c00 = z, c01 = z, c10 = z, c11 = z;
  for (int kt = 0; kt < 24; ++kt) {
    int k0 = kt * 32;
    const unsigned short* t0 = Wp + ((size_t)kt * 48 + (n0 >> 4)) * 512;
    if (kt + 1 < 24)
      __builtin_prefetch(Wp + ((size_t)(kt + 1) * 48 + (n0 >> 4)) * 512 + lane * 16, 0, 1);
    v8u a0 = load_afrag(A, DIM, m0,      k0, lane);
    v8u a1 = load_afrag(A, DIM, m0 + 16, k0, lane);
    v8u b0 = load_bfrag_packed(t0,       lane);
    v8u b1 = load_bfrag_packed(t0 + 512, lane);
    c00 = wmma_bf16(a0, b0, c00);
    c01 = wmma_bf16(a0, b1, c01);
    c10 = wmma_bf16(a1, b0, c10);
    c11 = wmma_bf16(a1, b1, c11);
  }
  int half = lane >> 4;
  int col0 = n0 + (lane & 15);
  float bb0 = bias[col0], bb1 = bias[col0 + 16];
#pragma unroll
  for (int j = 0; j < 8; ++j) {
    int r0 = m0 + j + 8 * half;
    int r1 = r0 + 16;
    C[(size_t)r0 * DIM + col0]      = f2bf(c00[j] + bb0);
    C[(size_t)r0 * DIM + col0 + 16] = f2bf(c01[j] + bb1);
    C[(size_t)r1 * DIM + col0]      = f2bf(c10[j] + bb0);
    C[(size_t)r1 * DIM + col0 + 16] = f2bf(c11[j] + bb1);
  }
}

// -------- trilinear: LDS-staged q/a tiles shared by 4 waves of a block --------
// Async global->LDS copy (CDNA5, tracked by ASYNCcnt), 16B per lane per issue.
__device__ __forceinline__ void async_copy_b128(void* lds_dst, const void* gsrc) {
  unsigned            lds = (unsigned)(unsigned long long)lds_dst;     // low 32b = LDS offset
  unsigned long long  ga  = (unsigned long long)gsrc;
  asm volatile("global_load_async_to_lds_b128 %0, %1, off"
               :: "v"(lds), "v"(ga) : "memory");
}

// Stage qh (32x64) and ah (32x64) bf16 tiles of one (b,h) into LDS; 128 threads.
__device__ __forceinline__ void stage_qa_tiles(const unsigned short* __restrict__ qp,
                                               const unsigned short* __restrict__ ap,
                                               int b, int h,
                                               unsigned short* qs, unsigned short* as_) {
  int t = threadIdx.x;                       // 0..127
#pragma unroll
  for (int i = 0; i < 4; ++i) {
    int c   = i * 128 + t;                   // 0..511 chunks of 16B
    int row = c >> 3, sub = (c & 7) * 8;     // 64 rows x 8 chunks, sub in elems
    if (row < NQq) {
      async_copy_b128(qs + row * HW + sub,
                      qp + ((size_t)(b * NQq + row)) * DIM + h * HW + sub);
    } else {
      int r = row - NQq;
      async_copy_b128(as_ + r * HW + sub,
                      ap + ((size_t)(b * NAa + r)) * DIM + h * HW + sub);
    }
  }
  asm volatile("s_wait_asynccnt 0x0" ::: "memory");
  __syncthreads();
}

// Score tiles for one (b,h,v): c[mt][nt] = (qh ⊙ vh)·ah^T * 1/sqrt(64); q/a from LDS
__device__ __forceinline__ void trilinear_tiles_lds(const unsigned short* qs,
                                                    const unsigned short* as_,
                                                    const unsigned short* vrow,
                                                    int lane, v8f c[2][2]) {
  v8f z = {};
  c[0][0] = z; c[0][1] = z; c[1][0] = z; c[1][1] = z;
#pragma unroll
  for (int k0 = 0; k0 < HW; k0 += 32) {
    v8u vf  = load_afrag_bcast(vrow, k0, lane);
    v8u vfs = bfscale(vf, 0.125f);                 // fold 1/sqrt(64), exact in bf16
    v8u q0  = load_afrag(qs, HW, 0,  k0, lane);
    v8u q1  = load_afrag(qs, HW, 16, k0, lane);
    v8u A0  = bfmul(q0, vfs);
    v8u A1  = bfmul(q1, vfs);
    v8u b0  = load_bfrag_rows(as_, HW, 0,  k0, lane);
    v8u b1  = load_bfrag_rows(as_, HW, 16, k0, lane);
    c[0][0] = wmma_bf16(A0, b0, c[0][0]);
    c[0][1] = wmma_bf16(A0, b1, c[0][1]);
    c[1][0] = wmma_bf16(A1, b0, c[1][0]);
    c[1][1] = wmma_bf16(A1, b1, c[1][1]);
  }
}

__global__ void init_stats_kernel(unsigned int* MK, float* Z, float* Sq, float* Sa) {
  int i = blockIdx.x * blockDim.x + threadIdx.x;
  if (i < BB * HEADS) { MK[i] = 0u; Z[i] = 0.0f; }
  if (i < BB * HEADS * NQq) Sq[i] = 0.0f;
  if (i < BB * HEADS * NAa) Sa[i] = 0.0f;
}

// Pass 1: per-(b,h) max of masked scores. Grid: (BB*HEADS)*25 blocks x 128 thr;
// block = one (b,h), 4 waves = 4 consecutive v (25*4 = NV exactly, all waves full).
__global__ void tri_max_kernel(const unsigned short* __restrict__ vp,
                               const unsigned short* __restrict__ qp,
                               const unsigned short* __restrict__ ap,
                               const int* __restrict__ vmask, const int* __restrict__ qmask,
                               const int* __restrict__ amask, unsigned int* __restrict__ MK) {
  __shared__ __align__(16) unsigned short qs[NQq * HW];
  __shared__ __align__(16) unsigned short as_[NAa * HW];
  int bh   = blockIdx.x / 25;
  int vblk = blockIdx.x % 25;
  int wave = threadIdx.x >> 5, lane = threadIdx.x & 31;
  int b = bh / HEADS, h = bh % HEADS;
  stage_qa_tiles(qp, ap, b, h, qs, as_);

  int v = vblk * 4 + wave;
  const unsigned short* vrow = vp + ((size_t)b * NVv + v) * DIM + h * HW;
  v8f c[2][2];
  trilinear_tiles_lds(qs, as_, vrow, lane, c);

  float vmb = -10000.0f * (float)vmask[b * NVv + v];
  const int* qm = qmask + b * NQq;
  const int* am = amask + b * NAa;
  int half = lane >> 4, la = lane & 15;
  float am0 = -10000.0f * (float)am[la];
  float am1 = -10000.0f * (float)am[16 + la];
  float mx = -3.4e38f;
#pragma unroll
  for (int mt = 0; mt < 2; ++mt)
#pragma unroll
    for (int j = 0; j < 8; ++j) {
      int q = mt * 16 + 8 * half + j;
      float qb = vmb - 10000.0f * (float)qm[q];
      mx = fmaxf(mx, c[mt][0][j] + qb + am0);
      mx = fmaxf(mx, c[mt][1][j] + qb + am1);
    }
  for (int m = 16; m >= 1; m >>= 1) mx = fmaxf(mx, __shfl_xor(mx, m, 32));
  if (lane == 0) atomicMax(&MK[bh], fkey(mx));
}

// Pass 2: E=exp(score-M); accumulate unnormalized marginals Sv (direct), Sq/Sa/Z (atomic)
__global__ void tri_sum_kernel(const unsigned short* __restrict__ vp,
                               const unsigned short* __restrict__ qp,
                               const unsigned short* __restrict__ ap,
                               const int* __restrict__ vmask, const int* __restrict__ qmask,
                               const int* __restrict__ amask, const unsigned int* __restrict__ MK,
                               float* __restrict__ Sv, float* __restrict__ Sq,
                               float* __restrict__ Sa, float* __restrict__ Z) {
  __shared__ __align__(16) unsigned short qs[NQq * HW];
  __shared__ __align__(16) unsigned short as_[NAa * HW];
  int bh   = blockIdx.x / 25;
  int vblk = blockIdx.x % 25;
  int wave = threadIdx.x >> 5, lane = threadIdx.x & 31;
  int b = bh / HEADS, h = bh % HEADS;
  stage_qa_tiles(qp, ap, b, h, qs, as_);

  int v = vblk * 4 + wave;
  const unsigned short* vrow = vp + ((size_t)b * NVv + v) * DIM + h * HW;
  v8f c[2][2];
  trilinear_tiles_lds(qs, as_, vrow, lane, c);

  float M = unfkey(MK[bh]);
  float vmb = -10000.0f * (float)vmask[b * NVv + v];
  const int* qm = qmask + b * NQq;
  const int* am = amask + b * NAa;
  int half = lane >> 4, la = lane & 15;
  float am0 = -10000.0f * (float)am[la];
  float am1 = -10000.0f * (float)am[16 + la];

  float e[2][2][8];
  float tot = 0.0f;
#pragma unroll
  for (int mt = 0; mt < 2; ++mt)
#pragma unroll
    for (int j = 0; j < 8; ++j) {
      int q = mt * 16 + 8 * half + j;
      float qb = vmb - 10000.0f * (float)qm[q];
      float e0 = __expf(c[mt][0][j] + qb + am0 - M);
      float e1 = __expf(c[mt][1][j] + qb + am1 - M);
      e[mt][0][j] = e0; e[mt][1][j] = e1;
      tot += e0 + e1;
    }
  // total over all 1024 score entries of this (b,h,v)
  float t = tot;
  for (int m = 16; m >= 1; m >>= 1) t += __shfl_xor(t, m, 32);
  if (lane == 0) { Sv[bh * NVv + v] = t; atomicAdd(&Z[bh], t); }

  // Sq[q]: sum over a (both nt tiles, both a-halves of the wave)
#pragma unroll
  for (int mt = 0; mt < 2; ++mt)
#pragma unroll
    for (int j = 0; j < 8; ++j) {
      float sq = e[mt][0][j] + e[mt][1][j];
      for (int m = 8; m >= 1; m >>= 1) sq += __shfl_xor(sq, m, 32);
      if (la == 0) atomicAdd(&Sq[bh * NQq + mt * 16 + 8 * half + j], sq);
    }
  // Sa[a]: sum over q (all 8 VGPRs, both mt tiles, both lane-halves)
#pragma unroll
  for (int nt = 0; nt < 2; ++nt) {
    float sa = 0.0f;
#pragma unroll
    for (int mt = 0; mt < 2; ++mt)
#pragma unroll
      for (int j = 0; j < 8; ++j) sa += e[mt][nt][j];
    sa += __shfl_xor(sa, 16, 32);
    if (half == 0) atomicAdd(&Sa[bh * NAa + nt * 16 + la], sa);
  }
}

// out = [v_out ; q_out ; a_out], each = proj_bf16 * marginal / Z
__global__ void finalize_kernel(const unsigned short* __restrict__ vo,
                                const unsigned short* __restrict__ qo,
                                const unsigned short* __restrict__ ao,
                                const float* __restrict__ Sv, const float* __restrict__ Sq,
                                const float* __restrict__ Sa, const float* __restrict__ Z,
                                float* __restrict__ out) {
  const int NVE = BB * NVv * DIM;   // 2457600
  const int NQE = BB * NQq * DIM;   // 786432
  int i = blockIdx.x * blockDim.x + threadIdx.x;
  if (i < NVE) {
    int b = i / (NVv * DIM), r = i % (NVv * DIM);
    int v = r / DIM, d = r % DIM, h = d >> 6;
    int bh = b * HEADS + h;
    out[i] = bf2f(vo[i]) * Sv[bh * NVv + v] / Z[bh];
  } else if (i < NVE + NQE) {
    int j = i - NVE;
    int b = j / (NQq * DIM), r = j % (NQq * DIM);
    int q = r / DIM, d = r % DIM, h = d >> 6;
    int bh = b * HEADS + h;
    out[i] = bf2f(qo[j]) * Sq[bh * NQq + q] / Z[bh];
  } else if (i < NVE + 2 * NQE) {
    int j = i - NVE - NQE;
    int b = j / (NAa * DIM), r = j % (NAa * DIM);
    int a = r / DIM, d = r % DIM, h = d >> 6;
    int bh = b * HEADS + h;
    out[i] = bf2f(ao[j]) * Sa[bh * NAa + a] / Z[bh];
  }
}

// ================= host launcher =================
extern "C" void kernel_launch(void* const* d_in, const int* in_sizes, int n_in,
                              void* d_out, int out_size, void* d_ws, size_t ws_size,
                              hipStream_t stream) {
  const float* v     = (const float*)d_in[0];
  const float* q     = (const float*)d_in[1];
  const float* a     = (const float*)d_in[2];
  const int*   vmask = (const int*)d_in[3];
  const int*   qmask = (const int*)d_in[4];
  const int*   amask = (const int*)d_in[5];
  const float* Wv  = (const float*)d_in[6];  const float* bv  = (const float*)d_in[7];
  const float* Wq  = (const float*)d_in[8];  const float* bq  = (const float*)d_in[9];
  const float* Wa  = (const float*)d_in[10]; const float* ba  = (const float*)d_in[11];
  const float* Wvo = (const float*)d_in[12]; const float* bvo = (const float*)d_in[13];
  const float* Wqo = (const float*)d_in[14]; const float* bqo = (const float*)d_in[15];
  const float* Wao = (const float*)d_in[16]; const float* bao = (const float*)d_in[17];

  char* ws = (char*)d_ws;
  size_t off = 0;
  auto take = [&](size_t bytes) -> char* {
    char* p = ws + off;
    off = (off + bytes + 255) & ~(size_t)255;
    return p;
  };

  unsigned short* pk  = (unsigned short*)take(6ull * DIM * DIM * 2);
  unsigned short* vbf = (unsigned short*)take((size_t)BB * NVv * DIM * 2);
  unsigned short* qbf = (unsigned short*)take((size_t)BB * NQq * DIM * 2);
  unsigned short* abf = (unsigned short*)take((size_t)BB * NAa * DIM * 2);
  unsigned short* vpb = (unsigned short*)take((size_t)BB * NVv * DIM * 2);
  unsigned short* qpb = (unsigned short*)take((size_t)BB * NQq * DIM * 2);
  unsigned short* apb = (unsigned short*)take((size_t)BB * NAa * DIM * 2);
  unsigned short* vob = (unsigned short*)take((size_t)BB * NVv * DIM * 2);
  unsigned short* qob = (unsigned short*)take((size_t)BB * NQq * DIM * 2);
  unsigned short* aob = (unsigned short*)take((size_t)BB * NAa * DIM * 2);
  float*        Sv = (float*)take((size_t)BB * HEADS * NVv * 4);
  float*        Sq = (float*)take((size_t)BB * HEADS * NQq * 4);
  float*        Sa = (float*)take((size_t)BB * HEADS * NAa * 4);
  float*        Z  = (float*)take((size_t)BB * HEADS * 4);
  unsigned int* MK = (unsigned int*)take((size_t)BB * HEADS * 4);

  // 1) pack all six weight matrices into WMMA B-fragment layout
  PackArgs pa;
  pa.W[0] = Wv; pa.W[1] = Wq; pa.W[2] = Wa; pa.W[3] = Wvo; pa.W[4] = Wqo; pa.W[5] = Wao;
  pa.dst = pk;
  pack_weights_kernel<<<(6 * 1152 * 32) / 256, 256, 0, stream>>>(pa);

  // 2) convert activations fp32 -> bf16
  int nve = BB * NVv * DIM, nqe = BB * NQq * DIM;
  cvt_f32_bf16_kernel<<<(nve + 255) / 256, 256, 0, stream>>>(v, vbf, nve);
  cvt_f32_bf16_kernel<<<(nqe + 255) / 256, 256, 0, stream>>>(q, qbf, nqe);
  cvt_f32_bf16_kernel<<<(nqe + 255) / 256, 256, 0, stream>>>(a, abf, nqe);

  // 3) six projection GEMMs on the WMMA pipe
  auto gemm = [&](const unsigned short* A, int mat, const float* bias, unsigned short* C, int M) {
    int waves = (M >> 5) * 24;
    gemm_bf16_kernel<<<(waves * 32) / 256, 256, 0, stream>>>(
        A, pk + (size_t)mat * DIM * DIM, bias, C, M);
  };
  gemm(vbf, 0, bv,  vpb, BB * NVv);
  gemm(qbf, 1, bq,  qpb, BB * NQq);
  gemm(abf, 2, ba,  apb, BB * NAa);
  gemm(vpb, 3, bvo, vob, BB * NVv);
  gemm(qpb, 4, bqo, qob, BB * NQq);
  gemm(apb, 5, bao, aob, BB * NAa);

  // 4) softmax statistics: init, max pass, exp+marginal pass
  init_stats_kernel<<<48, 256, 0, stream>>>(MK, Z, Sq, Sa);
  int tblocks = BB * HEADS * 25;   // one (b,h) per block, 4 waves = 4 v's, 25 blocks/(b,h)
  tri_max_kernel<<<tblocks, 128, 0, stream>>>(vpb, qpb, apb, vmask, qmask, amask, MK);
  tri_sum_kernel<<<tblocks, 128, 0, stream>>>(vpb, qpb, apb, vmask, qmask, amask, MK,
                                              Sv, Sq, Sa, Z);

  // 5) outputs = projected tensors scaled by normalized marginals
  int tot = nve + 2 * nqe;   // 4030464, exact multiple of 256
  finalize_kernel<<<tot / 256, 256, 0, stream>>>(vob, qob, aob, Sv, Sq, Sa, Z, (float*)d_out);
}
